// WeightedFunctions_45260365365323
// MI455X (gfx1250) — matrix operand order repaired and checked by reading, not compiled, not run on hardware
//
#include <hip/hip_runtime.h>

// MoE-style weighted expert combine, algebraically folded to per-f GEMMs:
//   Weff[f,o,i] = sum_n weights[f,n]*W[n,o,i];  out[b,f,:] = x[b,f,:] @ Weff[f]^T + beff[f]
// fp32 accuracy via bf16 hi/lo split (3x v_wmma_f32_16x16x32_bf16 per tile).
// Roofline: ~268MB HBM @23.3TB/s ~= 11.5us floor; bf16-split matrix math ~= 10us -> memory bound.

typedef __attribute__((ext_vector_type(16))) __bf16 bf16x16;
typedef __attribute__((ext_vector_type(8)))  __bf16 bf16x8;
typedef __attribute__((ext_vector_type(8)))  float  f32x8;
typedef __attribute__((ext_vector_type(4)))  float  f32x4;

#define BB   1024
#define FF   128
#define INN  256
#define OUTT 256
#define NFNE 8

union FragB { bf16x16 v; bf16x8 h[2]; };

__device__ __forceinline__ void split_bf16(float v, __bf16& hi, __bf16& lo) {
  hi = (__bf16)v;
  lo = (__bf16)(v - (float)hi);
}

// ---------------------------------------------------------------------------
// Phase 1: Weff[f,o,i] = sum_n weights[f,n] * W[n,o,i], split to bf16 hi/lo,
// stored in WMMA B-fragment order: per (f, n-tile, k-chunk): 32 lanes x 16 bf16,
// lane = (o&15) + 16*(k>=16), element j = k&15.
// ---------------------------------------------------------------------------
__global__ __launch_bounds__(256) void weff_pack_kernel(
    const float* __restrict__ W, const float* __restrict__ weights,
    __bf16* __restrict__ bhiW, __bf16* __restrict__ bloW) {
  const int g    = blockIdx.x * 256 + threadIdx.x;   // 1,048,576 threads total
  const int iblk = g & 31;
  const int o    = (g >> 5) & 255;
  const int f    = g >> 13;
  const int i0   = iblk * 8;

  float acc[8];
#pragma unroll
  for (int u = 0; u < 8; ++u) acc[u] = 0.f;
#pragma unroll
  for (int n = 0; n < NFNE; ++n) {
    const float wn = weights[f * NFNE + n];
    const float* wp = W + ((size_t)(n * OUTT + o)) * INN + i0;
    f32x4 a = *(const f32x4*)wp;
    f32x4 b = *(const f32x4*)(wp + 4);
#pragma unroll
    for (int u = 0; u < 4; ++u) { acc[u] += wn * a[u]; acc[4 + u] += wn * b[u]; }
  }

  bf16x8 hi8, lo8;
#pragma unroll
  for (int u = 0; u < 8; ++u) {
    __bf16 h, l2; split_bf16(acc[u], h, l2);
    hi8[u] = h; lo8[u] = l2;
  }

  const int nt = o >> 4, N = o & 15, kc = i0 >> 5, k0 = i0 & 31;
  const int l  = N + 16 * (k0 >> 4);
  const int jg = (k0 >> 3) & 1;
  const size_t off = ((size_t)((f * 16 + nt) * 8 + kc) * 32 + l) * 16 + (size_t)jg * 8;
  *(bf16x8*)(bhiW + off) = hi8;
  *(bf16x8*)(bloW + off) = lo8;
}

// ---------------------------------------------------------------------------
// Phase 2: per-f GEMM (1024x256) = x_f (1024x256) @ Weff_f^T (256x256) + beff.
// Block = 256 threads (8 waves): tile 128(M) x 64(N); wave tile 32x32.
// A (x) staged via LDS in WMMA A-fragment order with hi/lo split;
// B fragments loaded directly from L2-resident workspace.
// ---------------------------------------------------------------------------
__global__ __launch_bounds__(256) void moe_wmma_kernel(
    const float* __restrict__ x, const float* __restrict__ weights,
    const float* __restrict__ bias,
    const __bf16* __restrict__ bhiW, const __bf16* __restrict__ bloW,
    float* __restrict__ out) {
  __shared__ __attribute__((aligned(32))) bf16x8 Ahi[8 * 64];  // 8 m16-tiles * 32 lanes * 2 halves
  __shared__ __attribute__((aligned(32))) bf16x8 Alo[8 * 64];

  const int f     = blockIdx.z;
  const int mBase = blockIdx.y * 128;
  const int nBase = blockIdx.x * 64;
  const int t     = threadIdx.x;
  const int lane  = t & 31;
  const int wave  = t >> 5;
  const int waveM = wave & 3;   // 4 waves over M
  const int waveN = wave >> 2;  // 2 waves over N

  // staging: thread t handles row sm, 16 consecutive k starting at sk0
  const int sm   = t >> 1;          // 0..127
  const int sk0  = (t & 1) * 16;    // 0 or 16
  const int sm16 = sm >> 4, smi = sm & 15;
  const float* xrow = x + ((size_t)(mBase + sm) * FF + f) * INN + sk0;

  f32x4 xv[4];
#pragma unroll
  for (int q = 0; q < 4; ++q) xv[q] = *(const f32x4*)(xrow + q * 4);

  f32x8 acc[2][2];
#pragma unroll
  for (int a = 0; a < 2; ++a)
#pragma unroll
    for (int b = 0; b < 2; ++b)
#pragma unroll
      for (int r = 0; r < 8; ++r) acc[a][b][r] = 0.f;

  for (int kc = 0; kc < 8; ++kc) {
    // ---- stage chunk kc into LDS (fragment order, hi/lo split) ----
#pragma unroll
    for (int gph = 0; gph < 2; ++gph) {
      const int ks   = sk0 + gph * 8;
      const int half = (ks >> 4) & 1;
      const int jg   = (ks >> 3) & 1;
      const int li   = smi + 16 * half;
      bf16x8 hi8, lo8;
#pragma unroll
      for (int u = 0; u < 8; ++u) {
        const int e = gph * 8 + u;
        __bf16 h, l2; split_bf16(xv[e >> 2][e & 3], h, l2);
        hi8[u] = h; lo8[u] = l2;
      }
      const int slot = sm16 * 64 + li * 2 + jg;
      Ahi[slot] = hi8;
      Alo[slot] = lo8;
    }
    __syncthreads();

    // prefetch next chunk's x while WMMAs run
    if (kc < 7) {
#pragma unroll
      for (int q = 0; q < 4; ++q) xv[q] = *(const f32x4*)(xrow + (kc + 1) * 32 + q * 4);
    }

    // ---- B fragments straight from workspace (L2-resident) ----
    FragB bh[2], bl[2], ah[2], al[2];
#pragma unroll
    for (int tn = 0; tn < 2; ++tn) {
      const int nt = (nBase >> 4) + waveN * 2 + tn;
      const size_t off = ((size_t)((f * 16 + nt) * 8 + kc) * 32 + lane) * 16;
      bh[tn].h[0] = *(const bf16x8*)(bhiW + off);
      bh[tn].h[1] = *(const bf16x8*)(bhiW + off + 8);
      bl[tn].h[0] = *(const bf16x8*)(bloW + off);
      bl[tn].h[1] = *(const bf16x8*)(bloW + off + 8);
    }
    // ---- A fragments from LDS ----
#pragma unroll
    for (int tm = 0; tm < 2; ++tm) {
      const int tg = waveM * 2 + tm;
      ah[tm].h[0] = Ahi[tg * 64 + lane * 2];
      ah[tm].h[1] = Ahi[tg * 64 + lane * 2 + 1];
      al[tm].h[0] = Alo[tg * 64 + lane * 2];
      al[tm].h[1] = Alo[tg * 64 + lane * 2 + 1];
    }

    // ---- 12 WMMAs: hi*hi + lo*hi + hi*lo per 16x16 tile ----
#pragma unroll
    for (int tm = 0; tm < 2; ++tm)
#pragma unroll
      for (int tn = 0; tn < 2; ++tn) {
        acc[tm][tn] = __builtin_amdgcn_wmma_f32_16x16x32_bf16(
            false, ah[tm].v, false, bh[tn].v, (short)0, acc[tm][tn], false, false);
        acc[tm][tn] = __builtin_amdgcn_wmma_f32_16x16x32_bf16(
            false, al[tm].v, false, bh[tn].v, (short)0, acc[tm][tn], false, false);
        acc[tm][tn] = __builtin_amdgcn_wmma_f32_16x16x32_bf16(
            false, ah[tm].v, false, bl[tn].v, (short)0, acc[tm][tn], false, false);
      }
    __syncthreads();
  }

  // ---- epilogue: add folded bias, store C (layout: VGPR r -> M=r+8*(lane>=16), N=lane&15) ----
#pragma unroll
  for (int tn = 0; tn < 2; ++tn) {
    const int o = nBase + waveN * 32 + tn * 16 + (lane & 15);
    float bo = 0.f;
#pragma unroll
    for (int n = 0; n < NFNE; ++n) bo += weights[f * NFNE + n] * bias[n * OUTT + o];
#pragma unroll
    for (int tm = 0; tm < 2; ++tm) {
      const int m0 = mBase + waveM * 32 + tm * 16 + 8 * (lane >> 4);
#pragma unroll
      for (int r = 0; r < 8; ++r) {
        out[((size_t)(m0 + r) * FF + f) * OUTT + o] = acc[tm][tn][r] + bo;
      }
    }
  }
}

// ---------------------------------------------------------------------------
extern "C" void kernel_launch(void* const* d_in, const int* in_sizes, int n_in,
                              void* d_out, int out_size, void* d_ws, size_t ws_size,
                              hipStream_t stream) {
  (void)in_sizes; (void)n_in; (void)out_size; (void)ws_size;
  const float* x       = (const float*)d_in[0];  // (B,F,IN)
  const float* weights = (const float*)d_in[1];  // (F,NFN)
  const float* W       = (const float*)d_in[2];  // (NFN,OUT,IN)
  const float* bvec    = (const float*)d_in[3];  // (NFN,OUT)
  float* out = (float*)d_out;                    // (B,F,OUT)

  __bf16* wsBhi = (__bf16*)d_ws;                           // 16 MB
  __bf16* wsBlo = wsBhi + (size_t)FF * OUTT * INN;         // +16 MB (needs 32 MB ws)

  // Phase 1: fold experts into Weff, split bf16 hi/lo, fragment-order layout.
  weff_pack_kernel<<<dim3((FF * OUTT * INN / 8) / 256), dim3(256), 0, stream>>>(
      W, weights, wsBhi, wsBlo);

  // Phase 2: 128 GEMMs via WMMA. grid = (OUT/64, B/128, F).
  moe_wmma_kernel<<<dim3(OUTT / 64, BB / 128, FF), dim3(256), 0, stream>>>(
      x, weights, bvec, wsBhi, wsBlo, out);
}
